// LongCatImageTransformer2DModel_70437463655062
// MI455X (gfx1250) — matrix operand order repaired and checked by reading, not compiled, not run on hardware
//
#include <hip/hip_runtime.h>
#include <hip/hip_bf16.h>

// ---------------------------------------------------------------------------
// LongCat image-transformer attention block for MI455X (gfx1250, wave32).
// bf16 WMMA (v_wmma_f32_16x16x32_bf16) for all three matmul stages, f32
// accumulation, flash-style online-softmax attention.
// Round 2: batched vector loads per tile (one s_wait per group instead of
// per-load serialization), WMMA chains broken by alternating accumulators.
// ---------------------------------------------------------------------------

typedef __attribute__((ext_vector_type(16))) __bf16 v16bf;
typedef __attribute__((ext_vector_type(8)))  __bf16 v8bf;
typedef __attribute__((ext_vector_type(8)))  float  v8f;

#define S_TXT 512
#define S_IMG 2048
#define S_ALL 2560
#define DIM   3072
#define NH    24
#define HD    128
#define QKVN  9216        // 3*H*D
#define RHALF 64          // rope half-dim (8+28+28)
#define EPS   1e-6f
#define ATT_SCALE 0.08838834764831845f   // 1/sqrt(128)

static __device__ __forceinline__ v8f wmma_bf16(v16bf a, v16bf b, v8f c) {
  return __builtin_amdgcn_wmma_f32_16x16x32_bf16(false, a, false, b, (short)0, c,
                                                 false, false);
}

static __device__ __forceinline__ v16bf pack2(v8bf lo, v8bf hi) {
  v16bf r;
#pragma unroll
  for (int j = 0; j < 8; ++j) { r[j] = lo[j]; r[8 + j] = hi[j]; }
  return r;
}

static __device__ __forceinline__ v16bf cvt16(float4 x0, float4 x1,
                                              float4 x2, float4 x3) {
  v16bf r;
  r[0]  = (__bf16)x0.x; r[1]  = (__bf16)x0.y; r[2]  = (__bf16)x0.z; r[3]  = (__bf16)x0.w;
  r[4]  = (__bf16)x1.x; r[5]  = (__bf16)x1.y; r[6]  = (__bf16)x1.z; r[7]  = (__bf16)x1.w;
  r[8]  = (__bf16)x2.x; r[9]  = (__bf16)x2.y; r[10] = (__bf16)x2.z; r[11] = (__bf16)x2.w;
  r[12] = (__bf16)x3.x; r[13] = (__bf16)x3.y; r[14] = (__bf16)x3.z; r[15] = (__bf16)x3.w;
  return r;
}

// ---------------------------------------------------------------------------
// GEMM:  C[row_off+m, n] = sum_k A[m,k] * W[n,k]  + bias[n]
// A: f32 row-major [M,K], W: f32 row-major [N,K] (B = W^T is K-contiguous per
// output column).  All 20 b128 loads of a k-step are issued before any
// conversion/WMMA so they overlap in the memory pipeline.
// Block: 256 threads = 8 waves as 4(M) x 2(N): 64 x 128 block tile,
// each wave owns 16 x 64 via four 16x16 accumulators, K step = 32.
// ---------------------------------------------------------------------------
__global__ __launch_bounds__(256)
void gemm_xwt_bias(const float* __restrict__ A, const float* __restrict__ W,
                   const float* __restrict__ bias, float* __restrict__ C,
                   int K, int ldc, int row_off)
{
  const int lane   = threadIdx.x & 31;
  const int wave   = threadIdx.x >> 5;
  const int wm     = wave & 3;
  const int wn     = wave >> 2;
  const int l16    = lane & 15;
  const int laneHi = lane >> 4;

  const int m_base = blockIdx.y * 64 + wm * 16;
  const int n_base = blockIdx.x * 128 + wn * 64;

  const float* Arow = A + (size_t)(m_base + l16) * K + (laneHi ? 16 : 0);
  const float* Wr0  = W + (size_t)(n_base + l16) * K + (laneHi ? 16 : 0);
  const size_t wstep = (size_t)16 * K;   // 16 rows of W between nb blocks

  v8f acc[4];
#pragma unroll
  for (int nb = 0; nb < 4; ++nb)
#pragma unroll
    for (int i = 0; i < 8; ++i) acc[nb][i] = 0.0f;

#pragma unroll 2
  for (int k0 = 0; k0 < K; k0 += 32) {
    // ---- issue ALL loads for this k-step first (batched, one wait) ----
    const float* ap = Arow + k0;
    float4 a0 = *(const float4*)(ap);
    float4 a1 = *(const float4*)(ap + 4);
    float4 a2 = *(const float4*)(ap + 8);
    float4 a3 = *(const float4*)(ap + 12);

    float4 b00, b01, b02, b03, b10, b11, b12, b13;
    float4 b20, b21, b22, b23, b30, b31, b32, b33;
    {
      const float* wp0 = Wr0 + k0;
      const float* wp1 = wp0 + wstep;
      const float* wp2 = wp1 + wstep;
      const float* wp3 = wp2 + wstep;
      b00 = *(const float4*)(wp0);  b01 = *(const float4*)(wp0 + 4);
      b02 = *(const float4*)(wp0 + 8);  b03 = *(const float4*)(wp0 + 12);
      b10 = *(const float4*)(wp1);  b11 = *(const float4*)(wp1 + 4);
      b12 = *(const float4*)(wp1 + 8);  b13 = *(const float4*)(wp1 + 12);
      b20 = *(const float4*)(wp2);  b21 = *(const float4*)(wp2 + 4);
      b22 = *(const float4*)(wp2 + 8);  b23 = *(const float4*)(wp2 + 12);
      b30 = *(const float4*)(wp3);  b31 = *(const float4*)(wp3 + 4);
      b32 = *(const float4*)(wp3 + 8);  b33 = *(const float4*)(wp3 + 12);
    }
    if (k0 + 32 < K) __builtin_prefetch(Arow + k0 + 32, 0, 1);

    // ---- convert + WMMA ----
    const v16bf afrag = cvt16(a0, a1, a2, a3);
    const v16bf bf0 = cvt16(b00, b01, b02, b03);
    const v16bf bf1 = cvt16(b10, b11, b12, b13);
    const v16bf bf2 = cvt16(b20, b21, b22, b23);
    const v16bf bf3 = cvt16(b30, b31, b32, b33);
    acc[0] = wmma_bf16(afrag, bf0, acc[0]);
    acc[1] = wmma_bf16(afrag, bf1, acc[1]);
    acc[2] = wmma_bf16(afrag, bf2, acc[2]);
    acc[3] = wmma_bf16(afrag, bf3, acc[3]);
  }

  // ---- bias + store (C layout: row = i + 8*laneHi, col = l16) ----
#pragma unroll
  for (int nb = 0; nb < 4; ++nb) {
    const int col = n_base + nb * 16 + l16;
    const float b = bias[col];
#pragma unroll
    for (int i = 0; i < 8; ++i) {
      const int r = row_off + m_base + i + laneHi * 8;
      C[(size_t)r * ldc + col] = acc[nb][i] + b;
    }
  }
}

// ---------------------------------------------------------------------------
// Per (s, h): RMS norm of q/k over D=128, rotary embedding, then pack:
//   Qb,Kb: bf16 [H][S][D]   (row-major, fragment friendly for Q.K^T)
//   Vt   : bf16 [H][D][S]   (transposed so P.V B-fragments are contiguous)
// ---------------------------------------------------------------------------
__global__ __launch_bounds__(128)
void rms_rope_pack(const float* __restrict__ qkv,
                   const float* __restrict__ rope_cos,
                   const float* __restrict__ rope_sin,
                   const float* __restrict__ nq_w, const float* __restrict__ nk_w,
                   const float* __restrict__ naq_w, const float* __restrict__ nak_w,
                   __bf16* __restrict__ Qb, __bf16* __restrict__ Kb,
                   __bf16* __restrict__ Vt)
{
  const int s = blockIdx.x;
  const int h = blockIdx.y;
  const int d = threadIdx.x;

  const float* row = qkv + (size_t)s * QKVN;
  const float qx = row[h * HD + d];
  const float kx = row[DIM + h * HD + d];
  const float vx = row[2 * DIM + h * HD + d];

  __shared__ float red[128];
  __shared__ float sq[128];
  __shared__ float sk[128];

  red[d] = qx * qx; __syncthreads();
#pragma unroll
  for (int off = 64; off > 0; off >>= 1) {
    if (d < off) red[d] += red[d + off];
    __syncthreads();
  }
  const float qss = red[0]; __syncthreads();

  red[d] = kx * kx; __syncthreads();
#pragma unroll
  for (int off = 64; off > 0; off >>= 1) {
    if (d < off) red[d] += red[d + off];
    __syncthreads();
  }
  const float kss = red[0]; __syncthreads();

  const bool txt = (s < S_TXT);
  const float wq = (txt ? naq_w : nq_w)[d];
  const float wk = (txt ? nak_w : nk_w)[d];
  const float qn = qx * rsqrtf(qss * (1.0f / HD) + EPS) * wq;
  const float kn = kx * rsqrtf(kss * (1.0f / HD) + EPS) * wk;

  sq[d] = qn; sk[d] = kn;
  __syncthreads();

  const int i = d >> 1;
  const float c  = rope_cos[(size_t)s * RHALF + i];
  const float sn = rope_sin[(size_t)s * RHALF + i];
  float qo, ko;
  if ((d & 1) == 0) {
    qo = sq[d] * c - sq[d + 1] * sn;
    ko = sk[d] * c - sk[d + 1] * sn;
  } else {
    qo = sq[d] * c + sq[d - 1] * sn;
    ko = sk[d] * c + sk[d - 1] * sn;
  }

  Qb[((size_t)h * S_ALL + s) * HD + d] = (__bf16)qo;
  Kb[((size_t)h * S_ALL + s) * HD + d] = (__bf16)ko;
  Vt[((size_t)h * HD + d) * S_ALL + s] = (__bf16)vx;
}

// ---------------------------------------------------------------------------
// Flash attention: grid (S/128, H), 256 threads = 8 waves, each wave owns
// 16 q-rows with full D=128 f32 accumulation in registers (8 x v8f).
// Per 64-key tile: 16 WMMA for Q.K^T (nb processed in pairs, loads batched),
// online softmax (shfl_xor reductions inside 16-lane halves, matching the
// C-layout row mapping), P converted C-layout -> A-layout through a
// wave-private LDS tile, 16 WMMA for P.V with batched V loads.
// ---------------------------------------------------------------------------
__global__ __launch_bounds__(256)
void flash_attn(const __bf16* __restrict__ Qb, const __bf16* __restrict__ Kb,
                const __bf16* __restrict__ Vt, float* __restrict__ attn)
{
  const int lane   = threadIdx.x & 31;
  const int wave   = threadIdx.x >> 5;
  const int l16    = lane & 15;
  const int laneHi = lane >> 4;
  const int h      = blockIdx.y;

  const int q_row = blockIdx.x * 128 + wave * 16 + l16;
  const __bf16* Qrow = Qb + ((size_t)h * S_ALL + q_row) * HD;
  const __bf16* Kh   = Kb + (size_t)h * S_ALL * HD + (laneHi ? 16 : 0);
  const __bf16* Vh   = Vt + (size_t)h * HD * S_ALL + (laneHi ? 16 : 0);

  __shared__ __bf16 plds[8][16][72];   // per-wave P tile, padded rows

  // Q fragments: 4 chunks of 32 over d, kept resident in registers.
  v16bf qf[4];
#pragma unroll
  for (int kc = 0; kc < 4; ++kc) {
    const int base = kc * 32 + (laneHi ? 16 : 0);
    v8bf lo = *(const v8bf*)(Qrow + base);
    v8bf hi = *(const v8bf*)(Qrow + base + 8);
    qf[kc] = pack2(lo, hi);
  }

  float mrow[8], lrow[8];
  v8f oacc[8];
#pragma unroll
  for (int i = 0; i < 8; ++i) { mrow[i] = -3.0e38f; lrow[i] = 0.0f; }
#pragma unroll
  for (int db = 0; db < 8; ++db)
#pragma unroll
    for (int i = 0; i < 8; ++i) oacc[db][i] = 0.0f;

  for (int kt = 0; kt < S_ALL; kt += 64) {
    // ---- S = Q . K^T for this 16 x 64 tile, nb in pairs ----
    v8f sacc[4];
#pragma unroll
    for (int nb = 0; nb < 4; ++nb)
#pragma unroll
      for (int i = 0; i < 8; ++i) sacc[nb][i] = 0.0f;

#pragma unroll
    for (int np = 0; np < 2; ++np) {
      const int nb0 = np * 2, nb1 = np * 2 + 1;
      const __bf16* Kr0 = Kh + (size_t)(kt + nb0 * 16 + l16) * HD;
      const __bf16* Kr1 = Kh + (size_t)(kt + nb1 * 16 + l16) * HD;
      // batch all 16 loads for the two key column-blocks
      v8bf k0l0 = *(const v8bf*)(Kr0 + 0),   k0h0 = *(const v8bf*)(Kr0 + 8);
      v8bf k0l1 = *(const v8bf*)(Kr0 + 32),  k0h1 = *(const v8bf*)(Kr0 + 40);
      v8bf k0l2 = *(const v8bf*)(Kr0 + 64),  k0h2 = *(const v8bf*)(Kr0 + 72);
      v8bf k0l3 = *(const v8bf*)(Kr0 + 96),  k0h3 = *(const v8bf*)(Kr0 + 104);
      v8bf k1l0 = *(const v8bf*)(Kr1 + 0),   k1h0 = *(const v8bf*)(Kr1 + 8);
      v8bf k1l1 = *(const v8bf*)(Kr1 + 32),  k1h1 = *(const v8bf*)(Kr1 + 40);
      v8bf k1l2 = *(const v8bf*)(Kr1 + 64),  k1h2 = *(const v8bf*)(Kr1 + 72);
      v8bf k1l3 = *(const v8bf*)(Kr1 + 96),  k1h3 = *(const v8bf*)(Kr1 + 104);
      // alternate accumulators to break WMMA->WMMA RAW chains
      sacc[nb0] = wmma_bf16(qf[0], pack2(k0l0, k0h0), sacc[nb0]);
      sacc[nb1] = wmma_bf16(qf[0], pack2(k1l0, k1h0), sacc[nb1]);
      sacc[nb0] = wmma_bf16(qf[1], pack2(k0l1, k0h1), sacc[nb0]);
      sacc[nb1] = wmma_bf16(qf[1], pack2(k1l1, k1h1), sacc[nb1]);
      sacc[nb0] = wmma_bf16(qf[2], pack2(k0l2, k0h2), sacc[nb0]);
      sacc[nb1] = wmma_bf16(qf[2], pack2(k1l2, k1h2), sacc[nb1]);
      sacc[nb0] = wmma_bf16(qf[3], pack2(k0l3, k0h3), sacc[nb0]);
      sacc[nb1] = wmma_bf16(qf[3], pack2(k1l3, k1h3), sacc[nb1]);
    }

    // ---- online softmax update ----
    float alpha[8];
#pragma unroll
    for (int i = 0; i < 8; ++i) {
      float mx = fmaxf(fmaxf(sacc[0][i], sacc[1][i]),
                       fmaxf(sacc[2][i], sacc[3][i])) * ATT_SCALE;
      mx = fmaxf(mx, __shfl_xor(mx, 1, 32));
      mx = fmaxf(mx, __shfl_xor(mx, 2, 32));
      mx = fmaxf(mx, __shfl_xor(mx, 4, 32));
      mx = fmaxf(mx, __shfl_xor(mx, 8, 32));
      const float mnew = fmaxf(mrow[i], mx);
      alpha[i] = __expf(mrow[i] - mnew);
      float rs = 0.0f;
#pragma unroll
      for (int nb = 0; nb < 4; ++nb) {
        const float p = __expf(sacc[nb][i] * ATT_SCALE - mnew);
        sacc[nb][i] = p;
        rs += p;
      }
      rs += __shfl_xor(rs, 1, 32);
      rs += __shfl_xor(rs, 2, 32);
      rs += __shfl_xor(rs, 4, 32);
      rs += __shfl_xor(rs, 8, 32);
      lrow[i] = lrow[i] * alpha[i] + rs;
      mrow[i] = mnew;
    }
#pragma unroll
    for (int db = 0; db < 8; ++db)
#pragma unroll
      for (int i = 0; i < 8; ++i) oacc[db][i] *= alpha[i];

    // ---- P: C-layout -> A-layout via wave-private LDS ----
#pragma unroll
    for (int nb = 0; nb < 4; ++nb)
#pragma unroll
      for (int i = 0; i < 8; ++i)
        plds[wave][i + laneHi * 8][nb * 16 + l16] = (__bf16)sacc[nb][i];
    __builtin_amdgcn_wave_barrier();   // LDS is in-order per wave

    // ---- O += P . V (two key-chunks of 32, db in groups of 4) ----
#pragma unroll
    for (int c2 = 0; c2 < 2; ++c2) {
      const int pbase = c2 * 32 + (laneHi ? 16 : 0);
      v8bf plo = *(const v8bf*)(&plds[wave][l16][pbase]);
      v8bf phi = *(const v8bf*)(&plds[wave][l16][pbase + 8]);
      const v16bf pf = pack2(plo, phi);
#pragma unroll
      for (int dg = 0; dg < 2; ++dg) {
        const __bf16* Vr0 = Vh + (size_t)((dg * 4 + 0) * 16 + l16) * S_ALL + kt + c2 * 32;
        const __bf16* Vr1 = Vh + (size_t)((dg * 4 + 1) * 16 + l16) * S_ALL + kt + c2 * 32;
        const __bf16* Vr2 = Vh + (size_t)((dg * 4 + 2) * 16 + l16) * S_ALL + kt + c2 * 32;
        const __bf16* Vr3 = Vh + (size_t)((dg * 4 + 3) * 16 + l16) * S_ALL + kt + c2 * 32;
        v8bf v0l = *(const v8bf*)(Vr0), v0h = *(const v8bf*)(Vr0 + 8);
        v8bf v1l = *(const v8bf*)(Vr1), v1h = *(const v8bf*)(Vr1 + 8);
        v8bf v2l = *(const v8bf*)(Vr2), v2h = *(const v8bf*)(Vr2 + 8);
        v8bf v3l = *(const v8bf*)(Vr3), v3h = *(const v8bf*)(Vr3 + 8);
        oacc[dg * 4 + 0] = wmma_bf16(pf, pack2(v0l, v0h), oacc[dg * 4 + 0]);
        oacc[dg * 4 + 1] = wmma_bf16(pf, pack2(v1l, v1h), oacc[dg * 4 + 1]);
        oacc[dg * 4 + 2] = wmma_bf16(pf, pack2(v2l, v2h), oacc[dg * 4 + 2]);
        oacc[dg * 4 + 3] = wmma_bf16(pf, pack2(v3l, v3h), oacc[dg * 4 + 3]);
      }
    }
    __builtin_amdgcn_wave_barrier();
  }

  // ---- normalize and store f32 attention output [S, H*D] ----
#pragma unroll
  for (int db = 0; db < 8; ++db)
#pragma unroll
    for (int i = 0; i < 8; ++i) {
      const int row = blockIdx.x * 128 + wave * 16 + i + laneHi * 8;
      attn[(size_t)row * DIM + h * HD + db * 16 + l16] =
          oacc[db][i] * (1.0f / lrow[i]);
    }
}

// ---------------------------------------------------------------------------
extern "C" void kernel_launch(void* const* d_in, const int* in_sizes, int n_in,
                              void* d_out, int out_size, void* d_ws, size_t ws_size,
                              hipStream_t stream)
{
  (void)in_sizes; (void)n_in; (void)out_size; (void)ws_size;

  const float* hidden    = (const float*)d_in[0];
  const float* encoder   = (const float*)d_in[1];
  const float* rope_cos  = (const float*)d_in[2];
  const float* rope_sin  = (const float*)d_in[3];
  const float* w_qkv     = (const float*)d_in[4];
  const float* b_qkv     = (const float*)d_in[5];
  const float* w_add     = (const float*)d_in[6];
  const float* b_add     = (const float*)d_in[7];
  const float* nq_w      = (const float*)d_in[8];
  const float* nk_w      = (const float*)d_in[9];
  const float* naq_w     = (const float*)d_in[10];
  const float* nak_w     = (const float*)d_in[11];
  const float* w_out     = (const float*)d_in[12];
  const float* b_out     = (const float*)d_in[13];
  const float* w_add_out = (const float*)d_in[14];
  const float* b_add_out = (const float*)d_in[15];
  float* out = (float*)d_out;

  // workspace carve-up (~173 MB)
  char* ws = (char*)d_ws;
  float*  qkv  = (float*)ws;  ws += (size_t)S_ALL * QKVN * sizeof(float);
  __bf16* Qb   = (__bf16*)ws; ws += (size_t)NH * S_ALL * HD * sizeof(__bf16);
  __bf16* Kb   = (__bf16*)ws; ws += (size_t)NH * S_ALL * HD * sizeof(__bf16);
  __bf16* Vt   = (__bf16*)ws; ws += (size_t)NH * HD * S_ALL * sizeof(__bf16);
  float*  attn = (float*)ws;  ws += (size_t)S_ALL * DIM * sizeof(float);

  const dim3 blk(256);

  // 1) QKV projections: txt rows [0,512), img rows [512,2560) of qkv buffer
  gemm_xwt_bias<<<dim3(QKVN / 128, S_TXT / 64), blk, 0, stream>>>(
      encoder, w_add, b_add, qkv, DIM, QKVN, 0);
  gemm_xwt_bias<<<dim3(QKVN / 128, S_IMG / 64), blk, 0, stream>>>(
      hidden, w_qkv, b_qkv, qkv, DIM, QKVN, S_TXT);

  // 2) RMS norm + RoPE + bf16 packing (Q/K row-major, V transposed)
  rms_rope_pack<<<dim3(S_ALL, NH), dim3(128), 0, stream>>>(
      qkv, rope_cos, rope_sin, nq_w, nk_w, naq_w, nak_w, Qb, Kb, Vt);

  // 3) flash attention -> attn [S, H*D] f32
  flash_attn<<<dim3(S_ALL / 128, NH), blk, 0, stream>>>(Qb, Kb, Vt, attn);

  // 4) output projections: img part first in d_out, then txt part
  gemm_xwt_bias<<<dim3(DIM / 128, S_IMG / 64), blk, 0, stream>>>(
      attn + (size_t)S_TXT * DIM, w_out, b_out, out, DIM, DIM, 0);
  gemm_xwt_bias<<<dim3(DIM / 128, S_TXT / 64), blk, 0, stream>>>(
      attn, w_add_out, b_add_out, out + (size_t)S_IMG * DIM, DIM, DIM, 0);
}
